// DecoderTransformer_83210696392854
// MI455X (gfx1250) — matrix operand order repaired and checked by reading, not compiled, not run on hardware
//
#include <hip/hip_runtime.h>
#include <math.h>

// ---------------------------------------------------------------------------
// GPT decoder forward for gfx1250 (MI455X), bf16 WMMA everywhere.
// Shapes: L=8, H=1024, NH=16, HD=64, B=4, T=1024, FF=4096, V=32000, M=B*T=4096
// ---------------------------------------------------------------------------

typedef unsigned short bhalf; // bf16 bit pattern storage
typedef __attribute__((ext_vector_type(16))) __bf16 v16bf;
typedef __attribute__((ext_vector_type(8)))  float  v8f;
typedef __attribute__((ext_vector_type(16))) unsigned short v16u;
typedef __attribute__((ext_vector_type(8)))  unsigned short v8u;
typedef __attribute__((ext_vector_type(4)))  int v4i;

#define L_  8
#define H_  1024
#define NH_ 16
#define HD_ 64
#define B_  4
#define T_  1024
#define FF_ 4096
#define V_  32000
#define M_  (B_ * T_)   // 4096 rows

__device__ __forceinline__ bhalf f2bf(float f) {
    unsigned u = __builtin_bit_cast(unsigned, f);
    u += 0x7FFFu + ((u >> 16) & 1u);          // round-to-nearest-even
    return (bhalf)(u >> 16);
}

// A-matrix fragment (16x32 bf16): per lane, halves 0-7 at k0+(lane/16)*8,
// halves 8-15 at k0+16+(lane/16)*8 (two 16B contiguous loads).
__device__ __forceinline__ v16bf load_frag_a(const bhalf* p) {
    v8u lo = *(const v8u*)p;
    v8u hi = *(const v8u*)(p + 16);
    v16u u = __builtin_shufflevector(lo, hi, 0,1,2,3,4,5,6,7,8,9,10,11,12,13,14,15);
    return __builtin_bit_cast(v16bf, u);
}
// B-matrix fragment (32x16 bf16): per lane, halves 0-7 at k0+(lane/16)*8,
// halves 8-15 at k0+16+(lane/16)*8 (two 16B contiguous loads).
__device__ __forceinline__ v16bf load_frag_b(const bhalf* p) {
    v8u lo = *(const v8u*)p;
    v8u hi = *(const v8u*)(p + 16);
    v16u u = __builtin_shufflevector(lo, hi, 0,1,2,3,4,5,6,7,8,9,10,11,12,13,14,15);
    return __builtin_bit_cast(v16bf, u);
}

__device__ __forceinline__ v8f wmma_bf16(v16bf a, v16bf b, v8f c) {
    return __builtin_amdgcn_wmma_f32_16x16x32_bf16(
        false, a, false, b, (short)0, c, false, false);
}

// ---- CDNA5 async copy (global -> LDS, 16B per lane, tracked by ASYNCcnt) ---
__device__ __forceinline__ void async_copy_b128(const bhalf* g, bhalf* l) {
#if __has_builtin(__builtin_amdgcn_global_load_async_to_lds_b128)
    __builtin_amdgcn_global_load_async_to_lds_b128(
        (v4i*)const_cast<bhalf*>(g), (v4i*)l, 0, 0);
#else
    unsigned lds_off = (unsigned)(unsigned long long)l;  // LDS addr = low 32 bits
    asm volatile("global_load_async_to_lds_b128 %0, %1, off"
                 :: "v"(lds_off), "v"(g) : "memory");
#endif
}

__device__ __forceinline__ void wait_async0() {
#if __has_builtin(__builtin_amdgcn_s_wait_asynccnt)
    __builtin_amdgcn_s_wait_asynccnt(0);
#else
    asm volatile("s_wait_asynccnt 0x0" ::: "memory");
#endif
}

// ---------------------------------------------------------------------------
// Generic WMMA GEMM:  C[M,N] = A[M,K](bf16) x W[N,K]^T(bf16)  (+bias, epilogue)
// Block = 256 threads = 8 waves. Macro tile 128x64; each wave owns a 32x32
// region computed as a 2x2 grid of 16x16 WMMA tiles.
// Weight macro-tile (64 cols x 64 K = 8KB) is staged into a double-buffered
// LDS ring by GLOBAL_LOAD_ASYNC_TO_LDS_B128 (ASYNCcnt), shared by all 8 waves;
// the next chunk streams while the current one feeds 8 WMMAs per wave.
// MODE 0: Cf = val + bias                     (LM head, f32 out)
// MODE 1: Cb = bf16(gelu(val + bias))         (FF1)
// MODE 2: Cf += val + bias                    (Wo / FF2 residual accumulate)
// MODE 3: head-major Q/K store  [b,nh,t,hd]   (bf16, no bias)
// MODE 4: transposed V store    [b,nh,hd,t]   (bf16, no bias)
// ---------------------------------------------------------------------------
template <int MODE>
__global__ __launch_bounds__(256) void gemm_wmma(
    const bhalf* __restrict__ A, const bhalf* __restrict__ W,
    const float* __restrict__ bias, float* __restrict__ Cf,
    bhalf* __restrict__ Cb, int M, int N, int K)
{
    __shared__ __align__(16) bhalf bs[2][64 * 64];   // 2 x 8KB weight tiles

    const int tid  = threadIdx.x;
    const int lane = tid & 31;
    const int w    = tid >> 5;
    const int colB = blockIdx.x * 64;
    const int row0 = blockIdx.y * 128 + (w >> 1) * 32;
    const int col0 = colB + (w & 1) * 32;
    const int mr   = lane & 15;        // tile-local N index / A-frag row
    const int kg   = lane >> 4;        // lane half-group

    // async stage of one 64x64 weight chunk: thread -> (row nl, 16-half seg)
    const int nl  = tid >> 2;          // 0..63
    const int seg = (tid & 3) << 4;    // 0,16,32,48
    auto stage_b = [&](int kb, int buf) {
        const bhalf* g = W + (size_t)(colB + nl) * K + kb + seg;
        bhalf* lp = &bs[buf][nl * 64 + seg];
        async_copy_b128(g, lp);
        async_copy_b128(g + 8, lp + 8);
    };

    const bhalf* pa0 = A + (size_t)(row0 + mr) * K + (kg << 3);
    const bhalf* pa1 = pa0 + (size_t)16 * K;

    stage_b(0, 0);
    int buf = 0;

    v8f acc00 = {}, acc01 = {}, acc10 = {}, acc11 = {};
    for (int kb = 0; kb < K; kb += 64) {
        wait_async0();
        __syncthreads();                       // DMA writes visible to all waves
        if (kb + 64 < K) stage_b(kb + 64, buf ^ 1);
        __builtin_prefetch(pa0 + 256, 0, 1);   // keep A stream ahead (global_prefetch)
        __builtin_prefetch(pa1 + 256, 0, 1);
#pragma unroll
        for (int ks = 0; ks < 2; ks++) {
            v16bf a0 = load_frag_a(pa0); pa0 += 32;
            v16bf a1 = load_frag_a(pa1); pa1 += 32;
            const bhalf* lb = &bs[buf][(size_t)((w & 1) * 32 + mr) * 64 + ks * 32 + (kg << 3)];
            v16bf b0 = load_frag_b(lb);            // ds_load_b128 x2
            v16bf b1 = load_frag_b(lb + 16 * 64);
            acc00 = wmma_bf16(a0, b0, acc00);
            acc01 = wmma_bf16(a0, b1, acc01);
            acc10 = wmma_bf16(a1, b0, acc10);
            acc11 = wmma_bf16(a1, b1, acc11);
        }
        buf ^= 1;
    }

    auto store_tile = [&](v8f acc, int ti, int tj) {
#pragma unroll
        for (int v = 0; v < 8; v++) {
            const int rr = row0 + ti * 16 + v + (kg << 3); // C: M = v + 8*(lane/16)
            const int cc = col0 + tj * 16 + mr;            //    N = lane%16
            float val = acc[v];
            if (MODE == 0) {
                Cf[(size_t)rr * N + cc] = val + bias[cc];
            } else if (MODE == 1) {
                float x = val + bias[cc];
                x = 0.5f * x * (1.0f + erff(x * 0.70710678118f));   // exact GELU
                Cb[(size_t)rr * N + cc] = f2bf(x);
            } else if (MODE == 2) {
                Cf[(size_t)rr * N + cc] += val + bias[cc];
            } else if (MODE == 3) {
                int n = cc >> 6, d = cc & 63, bb = rr >> 10, t = rr & 1023;
                Cb[((size_t)(bb * NH_ + n) * T_ + t) * HD_ + d] = f2bf(val);
            } else {
                int n = cc >> 6, d = cc & 63, bb = rr >> 10, t = rr & 1023;
                Cb[((size_t)(bb * NH_ + n) * HD_ + d) * T_ + t] = f2bf(val);
            }
        }
    };
    store_tile(acc00, 0, 0);
    store_tile(acc01, 0, 1);
    store_tile(acc10, 1, 0);
    store_tile(acc11, 1, 1);
}

// ---------------------------------------------------------------------------
// Attention: one block per (16-query tile, batch*head). 256 threads = 8 waves.
// Phase 1: S = Q K^T * scale  (WMMA, causal tile skip) -> LDS f32 [16][1024]
// Phase 2: causal softmax in LDS; probs re-quantized to bf16 [16][1024]
// Phase 3: O = P V  (WMMA over LDS probs x transposed V), write concat-heads
// ---------------------------------------------------------------------------
__global__ __launch_bounds__(256) void attn_kernel(
    const bhalf* __restrict__ q, const bhalf* __restrict__ k,
    const bhalf* __restrict__ vT, bhalf* __restrict__ o, float scale)
{
    extern __shared__ char smem[];
    float* s    = (float*)smem;                                  // 16*1024 f32
    bhalf* p    = (bhalf*)(smem + 16 * 1024 * 4);                // 16*1024 bf16
    float* red  = (float*)(smem + 16 * 1024 * 4 + 16 * 1024 * 2);// 256 f32
    float* rowv = red + 256;                                     // 16 f32

    const int tid  = threadIdx.x;
    const int lane = tid & 31;
    const int w    = tid >> 5;
    const int mr   = lane & 15;
    const int kg   = lane >> 4;
    const int q0   = blockIdx.x * 16;
    const int bh   = blockIdx.y;

    const bhalf* qh = q  + (size_t)bh * T_ * HD_;
    const bhalf* kh = k  + (size_t)bh * T_ * HD_;
    const bhalf* vh = vT + (size_t)bh * HD_ * T_;

    // ---- Phase 1: scores -------------------------------------------------
    v16bf qa[2];
#pragma unroll
    for (int kk = 0; kk < 2; kk++)
        qa[kk] = load_frag_a(qh + (size_t)(q0 + mr) * HD_ + kk * 32 + (kg << 3));

    const int ntiles = blockIdx.x + 1;          // causal: only tiles with s0 <= q0
    for (int ct = w; ct < ntiles; ct += 8) {
        const int s0 = ct * 16;
        v8f acc = {};
#pragma unroll
        for (int kk = 0; kk < 2; kk++) {
            v16bf bfr = load_frag_b(kh + (size_t)(s0 + mr) * HD_ + kk * 32 + (kg << 3));
            acc = wmma_bf16(qa[kk], bfr, acc);
        }
#pragma unroll
        for (int v = 0; v < 8; v++)
            s[(v + (kg << 3)) * 1024 + s0 + mr] = acc[v] * scale;
    }
    __syncthreads();

    // ---- Phase 2: causal softmax ----------------------------------------
    const int r = tid >> 4, sub = tid & 15;
    const int valid = q0 + r + 1;
    int kmax = ((q0 + 16 + 31) >> 5) << 5;
    if (kmax > T_) kmax = T_;

    float mx = -3.4e38f;
    for (int j = sub; j < valid; j += 16) mx = fmaxf(mx, s[r * 1024 + j]);
    red[tid] = mx;
    __syncthreads();
    if (sub == 0) {
        float m2 = red[tid];
        for (int i = 1; i < 16; i++) m2 = fmaxf(m2, red[tid + i]);
        rowv[r] = m2;
    }
    __syncthreads();
    mx = rowv[r];
    float e = 0.0f;
    for (int j = sub; j < valid; j += 16) {
        float ex = __expf(s[r * 1024 + j] - mx);
        s[r * 1024 + j] = ex;
        e += ex;
    }
    red[tid] = e;
    __syncthreads();
    if (sub == 0) {
        float t2 = 0.0f;
        for (int i = 0; i < 16; i++) t2 += red[tid + i];
        rowv[r] = 1.0f / t2;
    }
    __syncthreads();
    const float inv = rowv[r];
    for (int j = sub; j < kmax; j += 16)
        p[r * 1024 + j] = (j < valid) ? f2bf(s[r * 1024 + j] * inv) : (bhalf)0;
    __syncthreads();

    // ---- Phase 3: O = P x V (waves 0..3, one 16-wide d-slice each) -------
    if (w < 4) {
        const int d0 = w * 16;
        v8f acc = {};
        for (int kb = 0; kb < kmax; kb += 32) {
            v16bf afr = load_frag_a(p + mr * 1024 + kb + (kg << 3));       // LDS
            v16bf bfr = load_frag_b(vh + (size_t)(d0 + mr) * T_ + kb + (kg << 3));
            acc = wmma_bf16(afr, bfr, acc);
        }
        const int bb = bh / NH_, hn = bh % NH_;
#pragma unroll
        for (int v = 0; v < 8; v++) {
            const int m = v + (kg << 3);
            o[((size_t)(bb * T_ + q0 + m)) * H_ + hn * HD_ + d0 + mr] = f2bf(acc[v]);
        }
    }
}

// ---------------------------------------------------------------------------
// LayerNorm(row of 1024 f32) -> bf16, one block per row
// ---------------------------------------------------------------------------
__global__ __launch_bounds__(256) void ln_bf16_kernel(
    const float* __restrict__ x, const float* __restrict__ g,
    const float* __restrict__ bta, bhalf* __restrict__ y)
{
    __shared__ float rs_[256], rq_[256];
    const int row = blockIdx.x;
    const float* xr = x + (size_t)row * H_;
    float vbuf[4], sum = 0.0f, sq = 0.0f;
#pragma unroll
    for (int i = 0; i < 4; i++) {
        float v = xr[threadIdx.x + i * 256];
        vbuf[i] = v; sum += v; sq += v * v;
    }
    rs_[threadIdx.x] = sum; rq_[threadIdx.x] = sq;
    __syncthreads();
    for (int off = 128; off > 0; off >>= 1) {
        if (threadIdx.x < off) {
            rs_[threadIdx.x] += rs_[threadIdx.x + off];
            rq_[threadIdx.x] += rq_[threadIdx.x + off];
        }
        __syncthreads();
    }
    const float mu  = rs_[0] * (1.0f / H_);
    const float var = rq_[0] * (1.0f / H_) - mu * mu;
    const float inv = rsqrtf(var + 1e-5f);
#pragma unroll
    for (int i = 0; i < 4; i++) {
        int c = threadIdx.x + i * 256;
        y[(size_t)row * H_ + c] = f2bf((vbuf[i] - mu) * inv * g[c] + bta[c]);
    }
}

__global__ __launch_bounds__(256) void embed_kernel(
    const int* __restrict__ xx, const float* __restrict__ emb,
    const float* __restrict__ pos, float* __restrict__ h)
{
    const int row = blockIdx.x;
    const int t = row & (T_ - 1);
    const int tok = xx[row];
#pragma unroll
    for (int i = 0; i < 4; i++) {
        int c = threadIdx.x + i * 256;
        h[(size_t)row * H_ + c] = emb[(size_t)tok * H_ + c] + pos[(size_t)t * H_ + c];
    }
}

__global__ __launch_bounds__(256) void f32_to_bf16_kernel(
    const float* __restrict__ in, bhalf* __restrict__ out, size_t n)
{
    size_t i = (size_t)blockIdx.x * blockDim.x + threadIdx.x;
    const size_t stride = (size_t)gridDim.x * blockDim.x;
    for (; i < n; i += stride) out[i] = f2bf(in[i]);
}

// ---------------------------------------------------------------------------
// Host orchestration
// ---------------------------------------------------------------------------
extern "C" void kernel_launch(void* const* d_in, const int* in_sizes, int n_in,
                              void* d_out, int out_size, void* d_ws, size_t ws_size,
                              hipStream_t stream)
{
    (void)in_sizes; (void)n_in; (void)out_size; (void)ws_size;

    const int*   x     = (const int*)  d_in[0];
    const float* emb   = (const float*)d_in[1];
    const float* pos   = (const float*)d_in[2];
    const float* ln1_g = (const float*)d_in[3];
    const float* ln1_b = (const float*)d_in[4];
    const float* wq    = (const float*)d_in[5];
    const float* wk    = (const float*)d_in[6];
    const float* wv    = (const float*)d_in[7];
    const float* wo    = (const float*)d_in[8];
    const float* bo    = (const float*)d_in[9];
    const float* ln2_g = (const float*)d_in[10];
    const float* ln2_b = (const float*)d_in[11];
    const float* w1    = (const float*)d_in[12];
    const float* b1    = (const float*)d_in[13];
    const float* w2    = (const float*)d_in[14];
    const float* b2    = (const float*)d_in[15];
    const float* lnf_g = (const float*)d_in[16];
    const float* lnf_b = (const float*)d_in[17];
    const float* w_lm  = (const float*)d_in[18];
    const float* b_lm  = (const float*)d_in[19];

    char* ws = (char*)d_ws;
    size_t off = 0;
    auto alloc = [&](size_t bytes) -> char* {
        char* ptr = ws + off;
        off = (off + bytes + 255) & ~(size_t)255;
        return ptr;
    };

    // bf16 weight pool
    bhalf* wq_bf = (bhalf*)alloc((size_t)L_ * H_ * H_ * 2);
    bhalf* wk_bf = (bhalf*)alloc((size_t)L_ * H_ * H_ * 2);
    bhalf* wv_bf = (bhalf*)alloc((size_t)L_ * H_ * H_ * 2);
    bhalf* wo_bf = (bhalf*)alloc((size_t)L_ * H_ * H_ * 2);
    bhalf* w1_bf = (bhalf*)alloc((size_t)L_ * FF_ * H_ * 2);
    bhalf* w2_bf = (bhalf*)alloc((size_t)L_ * H_ * FF_ * 2);
    bhalf* wl_bf = (bhalf*)alloc((size_t)V_ * H_ * 2);
    // activations
    float* h    = (float*)alloc((size_t)M_ * H_ * 4);
    bhalf* ybf  = (bhalf*)alloc((size_t)M_ * H_ * 2);
    bhalf* qb   = (bhalf*)alloc((size_t)M_ * H_ * 2);
    bhalf* kb   = (bhalf*)alloc((size_t)M_ * H_ * 2);
    bhalf* vtb  = (bhalf*)alloc((size_t)M_ * H_ * 2);
    bhalf* attb = (bhalf*)alloc((size_t)M_ * H_ * 2);
    bhalf* ffb  = (bhalf*)alloc((size_t)M_ * FF_ * 2);

    // weight conversion (recomputed each call: deterministic, capture-safe)
    f32_to_bf16_kernel<<<1024, 256, 0, stream>>>(wq,   wq_bf, (size_t)L_ * H_ * H_);
    f32_to_bf16_kernel<<<1024, 256, 0, stream>>>(wk,   wk_bf, (size_t)L_ * H_ * H_);
    f32_to_bf16_kernel<<<1024, 256, 0, stream>>>(wv,   wv_bf, (size_t)L_ * H_ * H_);
    f32_to_bf16_kernel<<<1024, 256, 0, stream>>>(wo,   wo_bf, (size_t)L_ * H_ * H_);
    f32_to_bf16_kernel<<<1024, 256, 0, stream>>>(w1,   w1_bf, (size_t)L_ * FF_ * H_);
    f32_to_bf16_kernel<<<1024, 256, 0, stream>>>(w2,   w2_bf, (size_t)L_ * H_ * FF_);
    f32_to_bf16_kernel<<<1024, 256, 0, stream>>>(w_lm, wl_bf, (size_t)V_ * H_);

    embed_kernel<<<M_, 256, 0, stream>>>(x, emb, pos, h);

    const float scale = 0.03125f;  // H^-0.5 (reference scales by hidden_dim)
    const size_t attn_smem = 16 * 1024 * 4 + 16 * 1024 * 2 + 256 * 4 + 16 * 4;
    const dim3 gH(H_ / 64, M_ / 128), gFF(FF_ / 64, M_ / 128), gV(V_ / 64, M_ / 128);

    for (int l = 0; l < L_; l++) {
        ln_bf16_kernel<<<M_, 256, 0, stream>>>(h, ln1_g + l * H_, ln1_b + l * H_, ybf);

        gemm_wmma<3><<<gH, 256, 0, stream>>>(ybf, wq_bf + (size_t)l * H_ * H_,
                                             nullptr, nullptr, qb, M_, H_, H_);
        gemm_wmma<3><<<gH, 256, 0, stream>>>(ybf, wk_bf + (size_t)l * H_ * H_,
                                             nullptr, nullptr, kb, M_, H_, H_);
        gemm_wmma<4><<<gH, 256, 0, stream>>>(ybf, wv_bf + (size_t)l * H_ * H_,
                                             nullptr, nullptr, vtb, M_, H_, H_);

        attn_kernel<<<dim3(T_ / 16, B_ * NH_), 256, attn_smem, stream>>>(
            qb, kb, vtb, attb, scale);

        gemm_wmma<2><<<gH, 256, 0, stream>>>(attb, wo_bf + (size_t)l * H_ * H_,
                                             bo + l * H_, h, nullptr, M_, H_, H_);

        ln_bf16_kernel<<<M_, 256, 0, stream>>>(h, ln2_g + l * H_, ln2_b + l * H_, ybf);

        gemm_wmma<1><<<gFF, 256, 0, stream>>>(ybf, w1_bf + (size_t)l * FF_ * H_,
                                              b1 + l * FF_, nullptr, ffb, M_, FF_, H_);
        gemm_wmma<2><<<gH, 256, 0, stream>>>(ffb, w2_bf + (size_t)l * H_ * FF_,
                                             b2 + l * H_, h, nullptr, M_, H_, FF_);
    }

    ln_bf16_kernel<<<M_, 256, 0, stream>>>(h, lnf_g, lnf_b, ybf);
    gemm_wmma<0><<<gV, 256, 0, stream>>>(ybf, wl_bf, b_lm, (float*)d_out,
                                         nullptr, M_, V_, H_);
}
